// HGNNConv_69123203662121
// MI455X (gfx1250) — matrix operand order repaired and checked by reading, not compiled, not run on hardware
//
#include <hip/hip_runtime.h>

#define N_NODES 100000
#define N_EDGES 20000
#define D 128

typedef __attribute__((ext_vector_type(16))) _Float16 v16h;
typedef __attribute__((ext_vector_type(8)))  float    v8f;

// ---------------- zero init (float4 stores) ----------------
__global__ void k_zero4(float4* __restrict__ p, int n4) {
    int i = blockIdx.x * blockDim.x + threadIdx.x;
    if (i < n4) p[i] = make_float4(0.f, 0.f, 0.f, 0.f);
}

// ---------------- degree accumulation ----------------
__global__ void k_degrees(const int* __restrict__ v_idx, const int* __restrict__ e_idx,
                          float* __restrict__ dv, float* __restrict__ de, int nnz) {
    int i = blockIdx.x * blockDim.x + threadIdx.x;
    if (i >= nnz) return;
    atomicAdd(&dv[v_idx[i]], 1.0f);
    atomicAdd(&de[e_idx[i]], 1.0f);
}

// dv <- dv>0 ? dv^-0.5 : 0 ; de <- de>0 ? 1/de : 0
__global__ void k_invdeg(float* __restrict__ dv, float* __restrict__ de) {
    int i = blockIdx.x * blockDim.x + threadIdx.x;
    if (i < N_NODES) {
        float d = dv[i];
        dv[i] = d > 0.f ? rsqrtf(d) : 0.f;
    } else if (i < N_NODES + N_EDGES) {
        int j = i - N_NODES;
        float d = de[j];
        de[j] = d > 0.f ? 1.f / d : 0.f;
    }
}

// ---------------- WMMA GEMM: Xs = (X @ W + b) * inv_sqrt_dv[row] ----------------
// Block = 256 threads = 8 waves; each wave owns one 16-row tile and all 8 col tiles.
__global__ void __launch_bounds__(256) k_gemm_scale(
    const float* __restrict__ X, const float* __restrict__ W,
    const float* __restrict__ bias, const float* __restrict__ inv_sqrt_dv,
    float* __restrict__ Xs) {
    // W transposed into LDS as f16: WT[n][k] = W[k][n]; pad row to 136 halves.
    __shared__ _Float16 WT[D][D + 8];
    const int tid = threadIdx.x;
    for (int idx = tid; idx < D * D; idx += 256) {
        int k = idx >> 7, n = idx & (D - 1);
        WT[n][k] = (_Float16)W[idx];
    }
    __syncthreads();

    const int wave = tid >> 5;
    const int rt = blockIdx.x * 8 + wave;          // 16-row tile index
    if (rt >= N_NODES / 16) return;                 // wave-uniform exit (EXEC stays all-1)
    const int lane = tid & 31;
    const int half = lane >> 4;                     // K half-select per ISA A layout
    const int l16  = lane & 15;

    // A fragments (16x32 f16 layout): lane<16 -> K {0..7,16..23}+ks*32, lane>=16 -> +8
    const float* xrow = X + (size_t)(rt * 16 + l16) * D;
    v16h a[4];
#pragma unroll
    for (int ks = 0; ks < 4; ++ks) {
        const int kb = ks * 32 + half * 8;
        float4 f0 = *(const float4*)(xrow + kb);
        float4 f1 = *(const float4*)(xrow + kb + 4);
        float4 f2 = *(const float4*)(xrow + kb + 16);
        float4 f3 = *(const float4*)(xrow + kb + 20);
        v16h t;
        t[0] = (_Float16)f0.x; t[1] = (_Float16)f0.y; t[2] = (_Float16)f0.z; t[3] = (_Float16)f0.w;
        t[4] = (_Float16)f1.x; t[5] = (_Float16)f1.y; t[6] = (_Float16)f1.z; t[7] = (_Float16)f1.w;
        t[8] = (_Float16)f2.x; t[9] = (_Float16)f2.y; t[10] = (_Float16)f2.z; t[11] = (_Float16)f2.w;
        t[12] = (_Float16)f3.x; t[13] = (_Float16)f3.y; t[14] = (_Float16)f3.z; t[15] = (_Float16)f3.w;
        a[ks] = t;
    }

    // per-output-row D_v^{-1/2} scales (D VGPR r maps to row r + 8*half)
    float sc[8];
#pragma unroll
    for (int r = 0; r < 8; ++r) sc[r] = inv_sqrt_dv[rt * 16 + half * 8 + r];

#pragma unroll
    for (int ct = 0; ct < 8; ++ct) {
        v8f c = {};
#pragma unroll
        for (int ks = 0; ks < 4; ++ks) {
            // B fragment (32x16): lane half selects K {0..7,16..23} / +8, N = ct*16+l16
            v16h bfrag;
            const _Float16* src = &WT[ct * 16 + l16][ks * 32 + half * 8];
#pragma unroll
            for (int e = 0; e < 16; ++e) bfrag[e] = src[e + (e >> 3) * 8];
            c = __builtin_amdgcn_wmma_f32_16x16x32_f16(
                    false, a[ks], false, bfrag, (short)0, c, false, false);
        }
        const int col = ct * 16 + l16;
        const float bc = bias[col];
#pragma unroll
        for (int r = 0; r < 8; ++r) {
            const int row = rt * 16 + half * 8 + r;
            Xs[(size_t)row * D + col] = (c[r] + bc) * sc[r];
        }
    }
}

// ---------------- sparse pass 1: Ye[e] += Xs[v]  (one wave per nnz) ----------------
__global__ void k_scatter_v2e(const float* __restrict__ Xs, const int* __restrict__ v_idx,
                              const int* __restrict__ e_idx, float* __restrict__ Ye, int nnz) {
    const int w = (blockIdx.x * blockDim.x + threadIdx.x) >> 5;
    if (w >= nnz) return;
    const int lane = threadIdx.x & 31;
    const int v = v_idx[w];
    const int e = e_idx[w];
    const float4 s = *(const float4*)(Xs + (size_t)v * D + lane * 4);
    float* dst = Ye + (size_t)e * D + lane * 4;
    atomicAdd(dst + 0, s.x);
    atomicAdd(dst + 1, s.y);
    atomicAdd(dst + 2, s.z);
    atomicAdd(dst + 3, s.w);
}

// ---------------- sparse pass 2: out[v] += Ye[e] * inv_de[e] ----------------
__global__ void k_scatter_e2v(const float* __restrict__ Ye, const float* __restrict__ inv_de,
                              const int* __restrict__ v_idx, const int* __restrict__ e_idx,
                              float* __restrict__ out, int nnz) {
    const int w = (blockIdx.x * blockDim.x + threadIdx.x) >> 5;
    if (w >= nnz) return;
    const int lane = threadIdx.x & 31;
    const int v = v_idx[w];
    const int e = e_idx[w];
    const float s = inv_de[e];
    const float4 y = *(const float4*)(Ye + (size_t)e * D + lane * 4);
    float* dst = out + (size_t)v * D + lane * 4;
    atomicAdd(dst + 0, y.x * s);
    atomicAdd(dst + 1, y.y * s);
    atomicAdd(dst + 2, y.z * s);
    atomicAdd(dst + 3, y.w * s);
}

// ---------------- final: out = relu(out * inv_sqrt_dv[row]) ----------------
__global__ void k_relu_scale(float* __restrict__ out, const float* __restrict__ inv_sqrt_dv) {
    int i = blockIdx.x * blockDim.x + threadIdx.x;
    if (i >= N_NODES * D) return;
    float v = out[i] * inv_sqrt_dv[i >> 7];
    out[i] = v > 0.f ? v : 0.f;
}

extern "C" void kernel_launch(void* const* d_in, const int* in_sizes, int n_in,
                              void* d_out, int out_size, void* d_ws, size_t ws_size,
                              hipStream_t stream) {
    const float* X     = (const float*)d_in[0];
    const float* W     = (const float*)d_in[1];
    const float* bias  = (const float*)d_in[2];
    const int*   v_idx = (const int*)d_in[3];
    const int*   e_idx = (const int*)d_in[4];
    float*       out   = (float*)d_out;
    const int nnz = in_sizes[3];

    // workspace layout (floats): [dv | de | Ye | Xs]
    float* ws = (float*)d_ws;
    float* dv = ws;                               // N_NODES
    float* de = dv + N_NODES;                     // N_EDGES
    float* Ye = de + N_EDGES;                     // N_EDGES * D
    float* Xs = Ye + (size_t)N_EDGES * D;         // N_NODES * D

    // zero dv+de+Ye (contiguous) and out
    {
        int n4 = (N_NODES + N_EDGES + N_EDGES * D) / 4;
        k_zero4<<<(n4 + 255) / 256, 256, 0, stream>>>((float4*)ws, n4);
        int m4 = (N_NODES * D) / 4;
        k_zero4<<<(m4 + 255) / 256, 256, 0, stream>>>((float4*)out, m4);
    }

    k_degrees<<<(nnz + 255) / 256, 256, 0, stream>>>(v_idx, e_idx, dv, de, nnz);
    k_invdeg<<<(N_NODES + N_EDGES + 255) / 256, 256, 0, stream>>>(dv, de);

    // WMMA GEMM + D_v^{-1/2} scaling: 6250 row tiles, 8 waves per block
    k_gemm_scale<<<(N_NODES / 16 + 7) / 8, 256, 0, stream>>>(X, W, bias, dv, Xs);

    // sparse passes: one wave per nonzero
    k_scatter_v2e<<<(nnz * 8 + 63) / 64 /* = nnz*32 threads / 256 */, 256, 0, stream>>>(
        Xs, v_idx, e_idx, Ye, nnz);
    k_scatter_e2v<<<(nnz * 8 + 63) / 64, 256, 0, stream>>>(Ye, de, v_idx, e_idx, out, nnz);

    k_relu_scale<<<(N_NODES * D + 255) / 256, 256, 0, stream>>>(out, dv);
}
